// EGNNModel_34591666602698
// MI455X (gfx1250) — matrix operand order repaired and checked by reading, not compiled, not run on hardware
//
#include <hip/hip_runtime.h>

// ============================================================================
// EGNN (2 layers) on MI455X / gfx1250 — fully fused f16-WMMA implementation.
//
// Roofline: ~170 GFLOP total, dominated by per-edge MLPs (message/coord/
// edge_emb). Irreducible HBM traffic ~50MB (few us at 23.3 TB/s) and all
// gather/scatter tables (h: <=25.6MB, m_acc: 12.8MB) fit in the 192MB L2,
// so the kernel is compute-bound -> cast every MLP as 16-row-tile GEMMs on
// v_wmma_f32_16x16x32_f16 (f32 accum).
//
// Round-1 change: A-matrix fragments are register-cached per GEMM (loadA +
// gemmA) instead of being re-read from LDS for every n-tile — drops A-side
// ds_load_b128 traffic ~8x on the big GEMMs (occupancy is LDS-limited to
// ~1 wave/SIMD, so DS-pipe pressure is the serialization risk).
//
// Dead code removed vs reference: after the final layer, h is unused, so
// layer-1's node MLP and its scatter-mean m_i are never computed.
//
// Packed per-edge input layout (K kept 32-aligned; weight rows permuted to
// match at pack time):   [ h_src(FIN) | h_dst(FIN) | edge_attr(64) | d_sq | 0 ]
// ============================================================================

#define N_NODES 50000
#define N_EDGES 800000

typedef _Float16 h8   __attribute__((ext_vector_type(8)));
typedef _Float16 v16h __attribute__((ext_vector_type(16)));
typedef float    v8f  __attribute__((ext_vector_type(8)));

__device__ __forceinline__ v8f zero8() {
  v8f z = {0.f,0.f,0.f,0.f,0.f,0.f,0.f,0.f};
  return z;
}
__device__ __forceinline__ h8 h8zero() {
  h8 z = {(_Float16)0,(_Float16)0,(_Float16)0,(_Float16)0,
          (_Float16)0,(_Float16)0,(_Float16)0,(_Float16)0};
  return z;
}

// CDNA5 split-counter fence for wave-local LDS producer->consumer ordering.
#define LDS_FENCE() asm volatile("s_wait_dscnt 0" ::: "memory")

// ---------------------------------------------------------------------------
// WMMA fragment loaders (CDNA5 16x16x32 f16 VGPR layouts, wave32).
// A (16xK tile, row m = lane%16): halves 0..7 -> K = kOff+8g+i,
//                                 halves 8..15 -> K = kOff+16+8g+(i-8).
// B (KxN weights, col n = nOff + lane%16): halves i -> same K as A.
// ---------------------------------------------------------------------------
__device__ __forceinline__ v16h ldsA_frag(const _Float16* A, int stride, int kOff) {
  int l = threadIdx.x & 31;
  int m = l & 15, g = l >> 4;
  const _Float16* p = A + m*stride + kOff + 8*g;
  union { v16h v; h8 h[2]; } u;
  u.h[0] = *(const h8*)(p);
  u.h[1] = *(const h8*)(p + 16);
  return u.v;
}

__device__ __forceinline__ v16h ldsB_frag(const _Float16* W, int ncols, int kOff, int nOff) {
  int l = threadIdx.x & 31;
  int g = l >> 4;
  const _Float16* p = W + (kOff + 8*g)*ncols + nOff + (l & 15);
  v16h v;
  #pragma unroll
  for (int i = 0; i < 8; ++i) {
    v[i]     = p[i*ncols];
    v[i + 8] = p[(16 + i)*ncols];
  }
  return v;
}

// Load all K/32 A-fragments of a 16-row tile into registers.
template<int K>
__device__ __forceinline__ void loadA(v16h* a, const _Float16* A, int sa) {
  #pragma unroll
  for (int kt = 0; kt < K/32; ++kt) a[kt] = ldsA_frag(A, sa, kt*32);
}

// One 16x16 output tile from register-cached A and LDS-resident B.
template<int K>
__device__ __forceinline__ v8f gemmA(const v16h* a, const _Float16* W, int sw, int nOff) {
  v8f acc = zero8();
  #pragma unroll
  for (int kt = 0; kt < K/32; ++kt) {
    v16h b = ldsB_frag(W, sw, kt*32, nOff);
    acc = __builtin_amdgcn_wmma_f32_16x16x32_f16(false, a[kt], false, b,
                                                 (short)0, acc, false, false);
  }
  return acc;
}

// C/D layout: lane holds column n = nOff + lane%16, rows m = r + 8*(lane/16).
__device__ __forceinline__ void store_hidden_relu(_Float16* H, int sh, int nOff,
                                                  v8f acc, const float* bias) {
  int l = threadIdx.x & 31; int g = l >> 4; int col = l & 15;
  float bb = bias[nOff + col];
  #pragma unroll
  for (int r = 0; r < 8; ++r) {
    float v = acc[r] + bb;
    v = v > 0.f ? v : 0.f;
    H[(r + 8*g)*sh + nOff + col] = (_Float16)v;
  }
}

__device__ __forceinline__ void copyW(_Float16* dst, const _Float16* src, int n, int nthreads) {
  for (int i = threadIdx.x; i < n/8; i += nthreads)
    ((h8*)dst)[i] = ((const h8*)src)[i];
}

// ---------------------------------------------------------------------------
// Fused per-edge layer: gaussian smear + edge_emb MLP + message MLP
// (+ optional scatter-mean accum) + coord MLP + position scatter-add.
// ---------------------------------------------------------------------------
template<int FIN, int NW, bool DO_MACC>
__global__ __launch_bounds__(NW*32) void edge_kernel(
    const int* __restrict__ src, const int* __restrict__ dst,
    const _Float16* __restrict__ h,    // [N][FIN] f16
    const float* __restrict__ p,       // current positions [N][3]
    const float* __restrict__ pos,     // initial positions [N][3]
    const _Float16* We1, const float* be1, const _Float16* We2, const float* be2,
    const _Float16* W1,  const float* b1,  const _Float16* W2,  const float* b2,
    const _Float16* Wc1, const float* bc1, const _Float16* Wc2, const float* bc2,
    float* __restrict__ m_acc,         // [N][FIN] (only if DO_MACC)
    float* __restrict__ p_out)         // [N][3]
{
  constexpr int KPAD = 2*FIN + 96;     // 2*FIN + 64 (edge_attr) + 1 (d_sq) -> pad32
  __shared__ alignas(16) _Float16 sW1 [KPAD*FIN];
  __shared__ alignas(16) _Float16 sW2 [FIN*FIN];
  __shared__ alignas(16) _Float16 sWc1[FIN*FIN];
  __shared__ alignas(16) _Float16 sWc2[FIN*16];
  __shared__ alignas(16) _Float16 sWe1[32*64];
  __shared__ alignas(16) _Float16 sWe2[64*64];
  __shared__ alignas(16) _Float16 sA[NW][16*KPAD];
  __shared__ alignas(16) _Float16 sH[NW][16*FIN];
  __shared__ alignas(16) _Float16 sG[NW][16*32];
  __shared__ float sDvec [NW][48];
  __shared__ float sInvd [NW][16];
  __shared__ float sDinit[NW][16];
  __shared__ int   sDst  [NW][16];

  const int NT = NW*32;
  copyW(sW1,  W1,  KPAD*FIN, NT);
  copyW(sW2,  W2,  FIN*FIN,  NT);
  copyW(sWc1, Wc1, FIN*FIN,  NT);
  copyW(sWc2, Wc2, FIN*16,   NT);
  copyW(sWe1, We1, 32*64,    NT);
  copyW(sWe2, We2, 64*64,    NT);
  __syncthreads();

  int wave = threadIdx.x >> 5, lane = threadIdx.x & 31;
  _Float16* A  = sA[wave];
  _Float16* Hd = sH[wave];
  _Float16* G  = sG[wave];

  // Zero A once: pad columns [2FIN+65, KPAD) must stay zero for all tiles.
  for (int i = lane; i < 16*KPAD/8; i += 32) ((h8*)A)[i] = h8zero();

  const float step  = 30.0f/31.0f;           // linspace(0,30,32) spacing
  const float coeff = -0.5f/(step*step);

  const int tiles = N_EDGES/16;              // 50000, exact
  for (int tile = blockIdx.x*NW + wave; tile < tiles; tile += gridDim.x*NW) {
    int e0 = tile*16;
    // --- per-edge scalars (lanes 0..15, one edge each) ---
    if (lane < 16) {
      int e = e0 + lane;
      int s = src[e], d = dst[e];
      sDst[wave][lane] = d;
      float dx = p[3*s+0]-p[3*d+0];
      float dy = p[3*s+1]-p[3*d+1];
      float dz = p[3*s+2]-p[3*d+2];
      A[lane*KPAD + 2*FIN + 64] = (_Float16)(dx*dx + dy*dy + dz*dz);   // d_ij^2
      float vx = pos[3*s+0]-pos[3*d+0];
      float vy = pos[3*s+1]-pos[3*d+1];
      float vz = pos[3*s+2]-pos[3*d+2];
      float dn = sqrtf(vx*vx + vy*vy + vz*vz);
      sDvec[wave][lane*3+0] = vx;
      sDvec[wave][lane*3+1] = vy;
      sDvec[wave][lane*3+2] = vz;
      sDinit[wave][lane] = dn;
      sInvd[wave][lane]  = 1.0f/dn;
    }
    // --- gather h[src] -> cols [0,FIN), h[dst] -> cols [FIN,2FIN) ---
    {
      constexpr int SEGS = FIN/4;            // 2*FIN/8 eight-half segments/row
      for (int t = lane; t < 16*SEGS; t += 32) {
        int row = t / SEGS, seg = t % SEGS;
        int e = e0 + row;
        int node = (seg < SEGS/2) ? src[e] : dst[e];
        int so   = (seg < SEGS/2) ? seg : seg - SEGS/2;
        *(h8*)(A + row*KPAD + seg*8) = *(const h8*)(h + (size_t)node*FIN + so*8);
      }
    }
    LDS_FENCE();
    // --- gaussian smearing of d_init (2 lanes per edge, 16 exps each) ---
    {
      int er = lane & 15, half = lane >> 4;
      float dn = sDinit[wave][er];
      #pragma unroll
      for (int j = 0; j < 16; ++j) {
        int c = half*16 + j;
        float t = dn - step*(float)c;
        G[er*32 + c] = (_Float16)__expf(coeff*t*t);
      }
    }
    LDS_FENCE();
    // --- edge_emb MLP: 32 -> 64 relu -> 64, result -> A cols [2FIN,2FIN+64) ---
    {
      v16h af[1];
      loadA<32>(af, G, 32);
      for (int n = 0; n < 64; n += 16)
        store_hidden_relu(Hd, 64, n, gemmA<32>(af, sWe1, 64, n), be1);
    }
    LDS_FENCE();
    {
      v16h af[2];
      loadA<64>(af, Hd, 64);
      int g = lane >> 4, col = lane & 15;
      for (int n = 0; n < 64; n += 16) {
        v8f acc = gemmA<64>(af, sWe2, 64, n);
        float bb = be2[n + col];
        #pragma unroll
        for (int r = 0; r < 8; ++r)
          A[(r + 8*g)*KPAD + 2*FIN + n + col] = (_Float16)(acc[r] + bb);
      }
    }
    LDS_FENCE();
    // --- message MLP l1: KPAD -> FIN relu ---
    {
      v16h af[KPAD/32];
      loadA<KPAD>(af, A, KPAD);
      for (int n = 0; n < FIN; n += 16)
        store_hidden_relu(Hd, FIN, n, gemmA<KPAD>(af, sW1, FIN, n), b1);
    }
    LDS_FENCE();
    // --- message MLP l2: FIN -> FIN; m_ij -> (optional) m_acc atomics + f16 in A ---
    {
      v16h af[FIN/32];
      loadA<FIN>(af, Hd, FIN);
      int g = lane >> 4, col = lane & 15;
      for (int n = 0; n < FIN; n += 16) {
        v8f acc = gemmA<FIN>(af, sW2, FIN, n);
        float bb = b2[n + col];
        #pragma unroll
        for (int r = 0; r < 8; ++r) {
          int row = r + 8*g;
          float v = acc[r] + bb;
          if (DO_MACC)
            atomicAdd(&m_acc[(size_t)sDst[wave][row]*FIN + n + col], v);
          A[row*KPAD + n + col] = (_Float16)v;   // m_ij (f16) reuses A cols [0,FIN)
        }
      }
    }
    LDS_FENCE();
    // --- coord MLP: FIN -> FIN relu -> 3 (N-padded to 16) ---
    {
      v16h af[FIN/32];
      loadA<FIN>(af, A, KPAD);
      for (int n = 0; n < FIN; n += 16)
        store_hidden_relu(Hd, FIN, n, gemmA<FIN>(af, sWc1, FIN, n), bc1);
    }
    LDS_FENCE();
    {
      v16h af[FIN/32];
      loadA<FIN>(af, Hd, FIN);
      v8f acc = gemmA<FIN>(af, sWc2, 16, 0);
      int g = lane >> 4, col = lane & 15;
      if (col < 3) {
        float bb = bc2[col];
        #pragma unroll
        for (int r = 0; r < 8; ++r) {
          int row = r + 8*g;
          float v = (acc[r] + bb) * sDvec[wave][row*3 + col] * sInvd[wave][row];
          atomicAdd(&p_out[(size_t)sDst[wave][row]*3 + col], v);
        }
      }
    }
  }
}

// ---------------------------------------------------------------------------
// node_emb: h0 = MLP(x), 32 -> 64 relu -> 64, f16 output.
// ---------------------------------------------------------------------------
__global__ __launch_bounds__(256) void node_emb_kernel(
    const float* __restrict__ x,
    const _Float16* W1, const float* b1,
    const _Float16* W2, const float* b2,
    _Float16* __restrict__ h0)
{
  __shared__ alignas(16) _Float16 sW1[32*64];
  __shared__ alignas(16) _Float16 sW2[64*64];
  __shared__ alignas(16) _Float16 sA[8][16*32];
  __shared__ alignas(16) _Float16 sH[8][16*64];
  copyW(sW1, W1, 32*64, 256);
  copyW(sW2, W2, 64*64, 256);
  __syncthreads();
  int wave = threadIdx.x >> 5, lane = threadIdx.x & 31;
  _Float16* A = sA[wave]; _Float16* Hd = sH[wave];
  for (int tile = blockIdx.x*8 + wave; tile < N_NODES/16; tile += gridDim.x*8) {
    int n0 = tile*16;
    {
      int row = lane >> 1, off = (lane & 1)*16;
      const float* xp = x + (size_t)(n0+row)*32 + off;
      alignas(16) _Float16 tmp[16];
      #pragma unroll
      for (int j = 0; j < 16; ++j) tmp[j] = (_Float16)xp[j];
      *(h8*)(A + row*32 + off)     = *(h8*)tmp;
      *(h8*)(A + row*32 + off + 8) = *(h8*)(tmp + 8);
    }
    LDS_FENCE();
    {
      v16h af[1];
      loadA<32>(af, A, 32);
      for (int n = 0; n < 64; n += 16)
        store_hidden_relu(Hd, 64, n, gemmA<32>(af, sW1, 64, n), b1);
    }
    LDS_FENCE();
    {
      v16h af[2];
      loadA<64>(af, Hd, 64);
      int g = lane >> 4, col = lane & 15;
      for (int n = 0; n < 64; n += 16) {
        v8f acc = gemmA<64>(af, sW2, 64, n);
        float bb = b2[n + col];
        #pragma unroll
        for (int r = 0; r < 8; ++r)
          h0[(size_t)(n0 + r + 8*g)*64 + n + col] = (_Float16)(acc[r] + bb);
      }
    }
  }
}

// ---------------------------------------------------------------------------
// Layer-0 node update: h1 = MLP([h0 | m_acc/deg]), 128 -> 64 relu -> 128.
// ---------------------------------------------------------------------------
__global__ __launch_bounds__(256) void node_upd_kernel(
    const _Float16* __restrict__ h0, const float* __restrict__ m_acc,
    const float* __restrict__ deg,
    const _Float16* W1, const float* b1,
    const _Float16* W2, const float* b2,
    _Float16* __restrict__ h1)
{
  __shared__ alignas(16) _Float16 sW1[128*64];
  __shared__ alignas(16) _Float16 sW2[64*128];
  __shared__ alignas(16) _Float16 sA[8][16*128];
  __shared__ alignas(16) _Float16 sH[8][16*64];
  copyW(sW1, W1, 128*64, 256);
  copyW(sW2, W2, 64*128, 256);
  __syncthreads();
  int wave = threadIdx.x >> 5, lane = threadIdx.x & 31;
  _Float16* A = sA[wave]; _Float16* Hd = sH[wave];
  for (int tile = blockIdx.x*8 + wave; tile < N_NODES/16; tile += gridDim.x*8) {
    int n0 = tile*16;
    for (int t = lane; t < 128; t += 32) {        // h0 -> cols [0,64)
      int row = t >> 3, seg = t & 7;
      *(h8*)(A + row*128 + seg*8) = *(const h8*)(h0 + (size_t)(n0+row)*64 + seg*8);
    }
    for (int t = lane; t < 128; t += 32) {        // m_i = m_acc/deg -> cols [64,128)
      int row = t >> 3, seg = t & 7;
      float rdeg = 1.0f / fmaxf(deg[n0+row], 1.0f);
      const float* mp = m_acc + (size_t)(n0+row)*64 + seg*8;
      alignas(16) _Float16 tmp[8];
      #pragma unroll
      for (int j = 0; j < 8; ++j) tmp[j] = (_Float16)(mp[j] * rdeg);
      *(h8*)(A + row*128 + 64 + seg*8) = *(h8*)tmp;
    }
    LDS_FENCE();
    {
      v16h af[4];
      loadA<128>(af, A, 128);
      for (int n = 0; n < 64; n += 16)
        store_hidden_relu(Hd, 64, n, gemmA<128>(af, sW1, 64, n), b1);
    }
    LDS_FENCE();
    {
      v16h af[2];
      loadA<64>(af, Hd, 64);
      int g = lane >> 4, col = lane & 15;
      for (int n = 0; n < 128; n += 16) {
        v8f acc = gemmA<64>(af, sW2, 128, n);
        float bb = b2[n + col];
        #pragma unroll
        for (int r = 0; r < 8; ++r)
          h1[(size_t)(n0 + r + 8*g)*128 + n + col] = (_Float16)(acc[r] + bb);
      }
    }
  }
}

// ---------------------------------------------------------------------------
// Weight pack: f32 [rows][cols] -> f16 [kpad][npad], zero-padded.
// mode 1 = message-l1 row permutation (reference concat order
// [h_src,h_dst,d_sq,edge_attr] -> packed [h_src,h_dst,edge_attr,d_sq]).
// ---------------------------------------------------------------------------
__global__ void pack_w_kernel(const float* __restrict__ src, _Float16* __restrict__ dst,
                              int rows, int cols, int kpad, int npad, int mode, int fin)
{
  int i = blockIdx.x*256 + threadIdx.x;
  if (i >= kpad*npad) return;
  int k = i / npad, n = i % npad;
  int sr;
  if (mode == 0) {
    sr = (k < rows) ? k : -1;
  } else {
    if      (k < 2*fin)       sr = k;         // h_src, h_dst rows
    else if (k < 2*fin + 64)  sr = k + 1;     // edge_attr rows (skip d_sq row)
    else if (k == 2*fin + 64) sr = 2*fin;     // d_sq row
    else sr = -1;                             // zero pad
  }
  float v = (sr >= 0 && n < cols) ? src[(size_t)sr*cols + n] : 0.f;
  dst[i] = (_Float16)v;
}

__global__ void deg_kernel(const int* __restrict__ dst, float* __restrict__ deg)
{
  int e = blockIdx.x*256 + threadIdx.x;
  if (e < N_EDGES) atomicAdd(&deg[dst[e]], 1.0f);
}

// ---------------------------------------------------------------------------
extern "C" void kernel_launch(void* const* d_in, const int* in_sizes, int n_in,
                              void* d_out, int out_size, void* d_ws, size_t ws_size,
                              hipStream_t stream)
{
  (void)in_sizes; (void)n_in; (void)ws_size;
  const float* x   = (const float*)d_in[0];
  const float* pos = (const float*)d_in[1];
  const int*   ei  = (const int*)d_in[2];
  const int* eSrc = ei;
  const int* eDst = ei + N_EDGES;

  // params flattened recursively in dict insertion order:
  // node_emb{l1{w,b},l2{w,b}}, edge_emb{...}, layers[0]{message,node,coord}, layers[1]{...}
  const float* ne_w1 = (const float*)d_in[3];
  const float* ne_b1 = (const float*)d_in[4];
  const float* ne_w2 = (const float*)d_in[5];
  const float* ne_b2 = (const float*)d_in[6];
  const float* ee_w1 = (const float*)d_in[7];
  const float* ee_b1 = (const float*)d_in[8];
  const float* ee_w2 = (const float*)d_in[9];
  const float* ee_b2 = (const float*)d_in[10];
  const float* m0_w1 = (const float*)d_in[11];
  const float* m0_b1 = (const float*)d_in[12];
  const float* m0_w2 = (const float*)d_in[13];
  const float* m0_b2 = (const float*)d_in[14];
  const float* n0_w1 = (const float*)d_in[15];
  const float* n0_b1 = (const float*)d_in[16];
  const float* n0_w2 = (const float*)d_in[17];
  const float* n0_b2 = (const float*)d_in[18];
  const float* c0_w1 = (const float*)d_in[19];
  const float* c0_b1 = (const float*)d_in[20];
  const float* c0_w2 = (const float*)d_in[21];
  const float* c0_b2 = (const float*)d_in[22];
  const float* m1_w1 = (const float*)d_in[23];
  const float* m1_b1 = (const float*)d_in[24];
  const float* m1_w2 = (const float*)d_in[25];
  const float* m1_b2 = (const float*)d_in[26];
  // d_in[27..30]: layer-1 node MLP — dead code (h unused after last layer)
  const float* c1_w1 = (const float*)d_in[31];
  const float* c1_b1 = (const float*)d_in[32];
  const float* c1_w2 = (const float*)d_in[33];
  const float* c1_b2 = (const float*)d_in[34];

  // --- workspace carve (~33 MB) ---
  char* base = (char*)d_ws;
  size_t off = 0;
  auto carve = [&](size_t bytes) -> char* {
    off = (off + 255) & ~(size_t)255;
    char* p = base + off;
    off += bytes;
    return p;
  };
  _Float16* h0   = (_Float16*)carve((size_t)N_NODES*64*2);
  _Float16* h1   = (_Float16*)carve((size_t)N_NODES*128*2);
  float*    mAcc = (float*)carve((size_t)N_NODES*64*4);
  float*    p1   = (float*)carve((size_t)N_NODES*3*4);
  float*    deg  = (float*)carve((size_t)N_NODES*4);
  _Float16* pwNE1 = (_Float16*)carve(32*64*2);
  _Float16* pwNE2 = (_Float16*)carve(64*64*2);
  _Float16* pwEE1 = (_Float16*)carve(32*64*2);
  _Float16* pwEE2 = (_Float16*)carve(64*64*2);
  _Float16* pwM0a = (_Float16*)carve(224*64*2);
  _Float16* pwM0b = (_Float16*)carve(64*64*2);
  _Float16* pwN0a = (_Float16*)carve(128*64*2);
  _Float16* pwN0b = (_Float16*)carve(64*128*2);
  _Float16* pwC0a = (_Float16*)carve(64*64*2);
  _Float16* pwC0b = (_Float16*)carve(64*16*2);
  _Float16* pwM1a = (_Float16*)carve(352*128*2);
  _Float16* pwM1b = (_Float16*)carve(128*128*2);
  _Float16* pwC1a = (_Float16*)carve(128*128*2);
  _Float16* pwC1b = (_Float16*)carve(128*16*2);

  hipMemsetAsync(mAcc,  0, (size_t)N_NODES*64*4, stream);
  hipMemsetAsync(p1,    0, (size_t)N_NODES*3*4, stream);
  hipMemsetAsync(deg,   0, (size_t)N_NODES*4, stream);
  hipMemsetAsync(d_out, 0, (size_t)out_size*4, stream);

  auto pack = [&](const float* s, _Float16* d, int rows, int cols,
                  int kpad, int npad, int mode, int fin) {
    int tot = kpad*npad;
    pack_w_kernel<<<dim3((tot+255)/256), dim3(256), 0, stream>>>(
        s, d, rows, cols, kpad, npad, mode, fin);
  };
  pack(ne_w1, pwNE1,  32,  64,  32,  64, 0, 0);
  pack(ne_w2, pwNE2,  64,  64,  64,  64, 0, 0);
  pack(ee_w1, pwEE1,  32,  64,  32,  64, 0, 0);
  pack(ee_w2, pwEE2,  64,  64,  64,  64, 0, 0);
  pack(m0_w1, pwM0a, 193,  64, 224,  64, 1, 64);
  pack(m0_w2, pwM0b,  64,  64,  64,  64, 0, 0);
  pack(n0_w1, pwN0a, 128,  64, 128,  64, 0, 0);
  pack(n0_w2, pwN0b,  64, 128,  64, 128, 0, 0);
  pack(c0_w1, pwC0a,  64,  64,  64,  64, 0, 0);
  pack(c0_w2, pwC0b,  64,   3,  64,  16, 0, 0);
  pack(m1_w1, pwM1a, 321, 128, 352, 128, 1, 128);
  pack(m1_w2, pwM1b, 128, 128, 128, 128, 0, 0);
  pack(c1_w1, pwC1a, 128, 128, 128, 128, 0, 0);
  pack(c1_w2, pwC1b, 128,   3, 128,  16, 0, 0);

  node_emb_kernel<<<dim3(512), dim3(256), 0, stream>>>(
      x, pwNE1, ne_b1, pwNE2, ne_b2, h0);
  deg_kernel<<<dim3((N_EDGES+255)/256), dim3(256), 0, stream>>>(eDst, deg);

  // Layer 0: message (with scatter-mean accum) + coord -> p1
  edge_kernel<64, 8, true><<<dim3(1024), dim3(256), 0, stream>>>(
      eSrc, eDst, h0, pos, pos,
      pwEE1, ee_b1, pwEE2, ee_b2,
      pwM0a, m0_b1, pwM0b, m0_b2,
      pwC0a, c0_b1, pwC0b, c0_b2,
      mAcc, p1);

  node_upd_kernel<<<dim3(512), dim3(256), 0, stream>>>(
      h0, mAcc, deg, pwN0a, n0_b1, pwN0b, n0_b2, h1);

  // Layer 1: message (no accum — dead) + coord -> d_out
  edge_kernel<128, 4, false><<<dim3(2048), dim3(128), 0, stream>>>(
      eSrc, eDst, h1, p1, pos,
      pwEE1, ee_b1, pwEE2, ee_b2,
      pwM1a, m1_b1, pwM1b, m1_b2,
      pwC1a, c1_b1, pwC1b, c1_b2,
      nullptr, (float*)d_out);
}
